// GlobalTransformer_7842610282854
// MI455X (gfx1250) — compile-verified, matrix operand order checked by reading, not run on hardware
//
#include <hip/hip_runtime.h>
#include <hip/hip_bf16.h>

// ---------------- problem constants ----------------
#define BB   4
#define NN   16384
#define CC   256
#define GRP  32
#define NP   512
#define HEADS 4
#define LAYERS 4
#define SL   48      // local seq padded (33 -> 48 rows, 64 K for A@V)
#define SGP  544     // global seq padded (513 -> 544 = 17*32)
#define RG   (BB*SGP) // 2176 global rows

typedef __attribute__((ext_vector_type(16))) _Float16 v16h;
typedef __attribute__((ext_vector_type(8)))  float    v8f;

// ---------------- WMMA helpers (ISA 7.12.2 layouts) ----------------
__device__ __forceinline__ v8f wmma32(v16h a, v16h b, v8f c) {
  return __builtin_amdgcn_wmma_f32_16x16x32_f16(false, a, false, b, (short)0, c, false, false);
}

// A fragment (16x32) or B fragment from K-contiguous storage: element (r,k) at p[r*ld + k].
// lane<16: row r0+lane, K {k0..k0+7, k0+16..k0+23}; lane>=16: row r0+lane-16, K {+8, +24}.
__device__ __forceinline__ v16h frag_row16(const _Float16* p, int ld, int r0, int k0, int lane) {
  int r  = r0 + (lane & 15);
  int kk = k0 + ((lane & 16) ? 8 : 0);
  const _Float16* s = p + r * ld + kk;
  v16h out;
  ((float4*)&out)[0] = *(const float4*)(s);
  ((float4*)&out)[1] = *(const float4*)(s + 16);
  return out;
}

// C tile (f32 16x16): lane<16 -> (m=v, n=lane); lane>=16 -> (m=v+8, n=lane-16)
__device__ __forceinline__ void tileStoreF16(_Float16* p, int ld, int m0, int n0, v8f c, int lane) {
  int n  = n0 + (lane & 15);
  int mb = m0 + ((lane & 16) ? 8 : 0);
#pragma unroll
  for (int v = 0; v < 8; ++v) p[(mb + v) * ld + n] = (_Float16)c[v];
}

// ---------------- weight convert ----------------
__global__ void k_cvt(const float* __restrict__ s, _Float16* __restrict__ d, int n) {
  for (int i = blockIdx.x * 256 + threadIdx.x; i < n; i += gridDim.x * 256)
    d[i] = (_Float16)s[i];
}

// ---------------- top-32 neighbor selection + centroids ----------------
// dynamic LDS layout (bytes): dist[16384]f @0 ; rv[256]f @65536 ; ri[256]i @66560 ;
// sel[32]i @67584 ; qp[4]f @67712 ; cpt[96]f @67728   -> pass 68352
__global__ void k_topk(const float* __restrict__ xyz, const int* __restrict__ fps,
                       const float* __restrict__ pdata, int* __restrict__ idx_out,
                       float* __restrict__ cent) {
  extern __shared__ __align__(16) char smem[];
  float* dist = (float*)smem;
  float* rv   = (float*)(smem + 65536);
  int*   ri   = (int*)(smem + 66560);
  int*   sel  = (int*)(smem + 67584);
  float* qp   = (float*)(smem + 67712);
  float* cpt  = (float*)(smem + 67728);

  int tid = threadIdx.x;
  int bp = blockIdx.x, b = bp >> 9;
  if (tid == 0) {
    int f = fps[bp];
    const float* q = xyz + (b * NN + f) * 3;
    qp[0] = q[0]; qp[1] = q[1]; qp[2] = q[2];
  }
  __syncthreads();
  float qx = qp[0], qy = qp[1], qz = qp[2];
  for (int n = tid; n < NN; n += 256) {
    const float* pr = xyz + (b * NN + n) * 3;
    float dx = pr[0] - qx, dy = pr[1] - qy, dz = pr[2] - qz;
    dist[n] = dx * dx + dy * dy + dz * dz;
  }
  __syncthreads();
  for (int g = 0; g < GRP; ++g) {
    float bv = 3.4e38f; int bi = 0x7fffffff;
    for (int n = tid; n < NN; n += 256) {
      float d = dist[n];
      if (d < bv || (d == bv && n < bi)) { bv = d; bi = n; }
    }
    rv[tid] = bv; ri[tid] = bi;
    __syncthreads();
    for (int s = 128; s > 0; s >>= 1) {
      if (tid < s) {
        if (rv[tid + s] < rv[tid] || (rv[tid + s] == rv[tid] && ri[tid + s] < ri[tid])) {
          rv[tid] = rv[tid + s]; ri[tid] = ri[tid + s];
        }
      }
      __syncthreads();
    }
    if (tid == 0) { sel[g] = ri[0]; dist[ri[0]] = 3.4e38f; }
    __syncthreads();
  }
  if (tid < GRP) {
    idx_out[bp * GRP + tid] = sel[tid];
    const float* pd = pdata + (size_t)(b * NN + sel[tid]) * CC;
    cpt[tid * 3 + 0] = pd[0]; cpt[tid * 3 + 1] = pd[1]; cpt[tid * 3 + 2] = pd[2];
  }
  __syncthreads();
  if (tid == 0) {
    float s0 = 0, s1 = 0, s2 = 0;
    for (int g = 0; g < GRP; ++g) { s0 += cpt[g*3]; s1 += cpt[g*3+1]; s2 += cpt[g*3+2]; }
    cent[bp * 3 + 0] = s0 * (1.f/GRP); cent[bp * 3 + 1] = s1 * (1.f/GRP); cent[bp * 3 + 2] = s2 * (1.f/GRP);
  }
}

// ---------------- closest centroid per point ----------------
__global__ void k_closest(const float* __restrict__ xyz, const float* __restrict__ cent,
                          int* __restrict__ closest) {
  __shared__ float ce[NP * 3];
  int b = blockIdx.y;
  for (int e = threadIdx.x; e < NP * 3; e += 256) ce[e] = cent[b * NP * 3 + e];
  __syncthreads();
  int n = blockIdx.x * 256 + threadIdx.x;
  const float* pr = xyz + (b * NN + n) * 3;
  float px = pr[0], py = pr[1], pz = pr[2];
  float bv = 3.4e38f; int bi = 0;
  for (int p = 0; p < NP; ++p) {
    float dx = px - ce[p*3], dy = py - ce[p*3+1], dz = pz - ce[p*3+2];
    float d = dx*dx + dy*dy + dz*dz;
    if (d < bv) { bv = d; bi = p; }
  }
  closest[b * NN + n] = bi;
}

// ---------------- scatter counts ----------------
__global__ void k_cnt(const int* __restrict__ idx, float* __restrict__ cnt) {
  int e = blockIdx.x * 256 + threadIdx.x;   // 0..65535
  int b = e >> 14;
  atomicAdd(&cnt[b * NN + idx[e]], 1.0f);
}

// ---------------- per-row LayerNorm on LDS residual ----------------
__device__ __forceinline__ void row_ln_lds(const float* xb, _Float16* nb,
                                           const float* w, const float* bia, int t) {
  const float* r = xb + t * CC;
  float m = 0.f;
  for (int c = 0; c < CC; ++c) m += r[c];
  m *= (1.f / CC);
  float v = 0.f;
  for (int c = 0; c < CC; ++c) { float d = r[c] - m; v += d * d; }
  v *= (1.f / CC);
  float rs = rsqrtf(v + 1e-5f);
  _Float16* dst = nb + t * CC;
  for (int c = 0; c < CC; ++c) dst[c] = (_Float16)((r[c] - m) * rs * w[c] + bia[c]);
}

// ---------------- local transformer mega-kernel ----------------
// dynamic LDS: xb 48x256 f32 @0 ; nb 48x256 f16 @49152 ; qk 48x768 f16 @73728 ;
// sb 4x48x48 f32 @147456 ; ab 4x48x64 f16 @184320 ; vt 4x64dx64t f16 @208896
// -> total 241664 bytes
#define XB_OFF 0
#define NB_OFF 49152
#define QK_OFF 73728
#define SB_OFF 147456
#define AB_OFF 184320
#define VT_OFF 208896
#define LOC_SMEM 241664

__global__ void k_local(const float* __restrict__ pdata, const int* __restrict__ idx,
                        const _Float16* __restrict__ qkvw, const _Float16* __restrict__ outw,
                        const _Float16* __restrict__ ffnw,
                        const float* ln1w, const float* ln1b,
                        const float* qkvb, const float* outb,
                        const float* ln2w, const float* ln2b, const float* ffnb,
                        const float* fnw, const float* fnb,
                        float* __restrict__ out, float* __restrict__ cfx) {
  extern __shared__ __align__(16) char smem[];
  float*    xb = (float*)(smem + XB_OFF);
  _Float16* nb = (_Float16*)(smem + NB_OFF);
  _Float16* qk = (_Float16*)(smem + QK_OFF);   // Q,K only (cols 0..511), 48 rows
  float*    sb = (float*)(smem + SB_OFF);
  _Float16* ab = (_Float16*)(smem + AB_OFF);
  _Float16* vt = (_Float16*)(smem + VT_OFF);   // V transposed: [head][d 0..63][token 0..63]

  const int tid = threadIdx.x, lane = tid & 31, wv = tid >> 5;
  const int bp = blockIdx.x, b = bp >> 9, p = bp & 511;
  const int* sel = idx + bp * GRP;

  // gather tokens: 0..31 grouped points, 32 zero token, 33..47 padding
  for (int e = tid; e < SL * CC; e += 256) {
    int t = e >> 8, c = e & 255;
    float v = 0.f;
    if (t < GRP) v = pdata[(size_t)(b * NN + sel[t]) * CC + c];
    xb[e] = v;
  }
  // zero vt token columns 48..63 (K padding for A@V); rows = 4 heads * 64 d = 256
  for (int e = tid; e < 256 * 16; e += 256) vt[(e >> 4) * 64 + 48 + (e & 15)] = (_Float16)0.f;
  __syncthreads();

  for (int l = 0; l < LAYERS; ++l) {
    const _Float16* Wq = qkvw + l * 768 * CC;
    const _Float16* Wo = outw + l * CC * CC;
    const _Float16* Wf = ffnw + l * CC * CC;

    // LN1 -> nb
    if (tid < SL) row_ln_lds(xb, nb, ln1w + l * CC, ln1b + l * CC, tid);
    __syncthreads();

    // QKV GEMM: (48x256)x(256x768); Q,K -> qk ; V -> vt (transposed at store)
    for (int t = wv; t < 144; t += 8) {
      int rt = t / 48, ct = t % 48;
      v8f acc = {};
#pragma unroll
      for (int kb = 0; kb < 8; ++kb) {
        v16h a  = frag_row16(nb, CC, rt * 16, kb * 32, lane);
        v16h bb = frag_row16(Wq, CC, ct * 16, kb * 32, lane);
        acc = wmma32(a, bb, acc);
      }
      int n = ct * 16 + (lane & 15), mb = rt * 16 + ((lane & 16) ? 8 : 0);
      float bv = qkvb[l * 768 + n];
      if (ct < 32) {
#pragma unroll
        for (int v = 0; v < 8; ++v) qk[(mb + v) * 768 + n] = (_Float16)(acc[v] + bv);
      } else {
        int hh = (ct - 32) >> 2;
        int dl = ((ct - 32) & 3) * 16 + (lane & 15);
        _Float16* vp = vt + hh * 4096 + dl * 64;   // [d][token]
#pragma unroll
        for (int v = 0; v < 8; ++v) vp[mb + v] = (_Float16)(acc[v] + bv);
      }
    }
    __syncthreads();

    // scores = Q K^T / 8 : 4 heads x 3x3 tiles
    for (int t = wv; t < 36; t += 8) {
      int h = t / 9, rr = t % 9, rt = rr / 3, ct = rr % 3;
      v8f acc = {};
#pragma unroll
      for (int kb = 0; kb < 2; ++kb) {
        v16h a  = frag_row16(qk + h * 64,       768, rt * 16, kb * 32, lane);
        v16h bb = frag_row16(qk + 256 + h * 64, 768, ct * 16, kb * 32, lane);
        acc = wmma32(a, bb, acc);
      }
      int n = ct * 16 + (lane & 15), mb = rt * 16 + ((lane & 16) ? 8 : 0);
#pragma unroll
      for (int v = 0; v < 8; ++v) sb[h * 2304 + (mb + v) * 48 + n] = acc[v] * 0.125f;
    }
    __syncthreads();

    // masked softmax over 33 valid keys -> ab (f16, 64-wide, zero padded)
    if (tid < 4 * SL) {
      int h = tid / SL, r = tid % SL;
      const float* srow = sb + h * 2304 + r * 48;
      float m = -1e30f;
      for (int j = 0; j < 33; ++j) m = fmaxf(m, srow[j]);
      float s = 0.f;
      for (int j = 0; j < 33; ++j) s += __expf(srow[j] - m);
      float inv = 1.f / s;
      _Float16* arow = ab + h * SL * 64 + r * 64;
      for (int j = 0; j < 33; ++j) arow[j] = (_Float16)(__expf(srow[j] - m) * inv);
      for (int j = 33; j < 64; ++j) arow[j] = (_Float16)0.f;
    }
    __syncthreads();

    // O = A V : 4 heads x 3 row-tiles x 4 col-tiles -> nb f16 (B from vt, K-contiguous)
    for (int t = wv; t < 48; t += 8) {
      int h = t / 12, rr = t % 12, rt = rr / 4, ct = rr % 4;
      v8f acc = {};
#pragma unroll
      for (int kb = 0; kb < 2; ++kb) {
        v16h a  = frag_row16(ab + h * SL * 64, 64, rt * 16, kb * 32, lane);
        v16h bb = frag_row16(vt + h * 4096, 64, ct * 16, kb * 32, lane);
        acc = wmma32(a, bb, acc);
      }
      tileStoreF16(nb, CC, rt * 16, h * 64 + ct * 16, acc, lane);
    }
    __syncthreads();

    // out-proj residual: xb += O Wo^T + bo : 3x16 tiles
    for (int t = wv; t < 48; t += 8) {
      int rt = t / 16, ct = t % 16;
      v8f acc = {};
#pragma unroll
      for (int kb = 0; kb < 8; ++kb) {
        v16h a  = frag_row16(nb, CC, rt * 16, kb * 32, lane);
        v16h bb = frag_row16(Wo, CC, ct * 16, kb * 32, lane);
        acc = wmma32(a, bb, acc);
      }
      int n = ct * 16 + (lane & 15), mb = rt * 16 + ((lane & 16) ? 8 : 0);
      float bv = outb[l * CC + n];
#pragma unroll
      for (int v = 0; v < 8; ++v) xb[(mb + v) * CC + n] += acc[v] + bv;
    }
    __syncthreads();

    // LN2 -> nb
    if (tid < SL) row_ln_lds(xb, nb, ln2w + l * CC, ln2b + l * CC, tid);
    __syncthreads();

    // FFN residual: xb += nb Wf^T + bf
    for (int t = wv; t < 48; t += 8) {
      int rt = t / 16, ct = t % 16;
      v8f acc = {};
#pragma unroll
      for (int kb = 0; kb < 8; ++kb) {
        v16h a  = frag_row16(nb, CC, rt * 16, kb * 32, lane);
        v16h bb = frag_row16(Wf, CC, ct * 16, kb * 32, lane);
        acc = wmma32(a, bb, acc);
      }
      int n = ct * 16 + (lane & 15), mb = rt * 16 + ((lane & 16) ? 8 : 0);
      float bv = ffnb[l * CC + n];
#pragma unroll
      for (int v = 0; v < 8; ++v) xb[(mb + v) * CC + n] += acc[v] + bv;
    }
    __syncthreads();
  }

  // final LN; scatter pts into out[...,:256] (pre-zeroed), token 32 -> global stack
  if (tid < 33) {
    const float* r = xb + tid * CC;
    float m = 0.f;
    for (int c = 0; c < CC; ++c) m += r[c];
    m *= (1.f / CC);
    float v = 0.f;
    for (int c = 0; c < CC; ++c) { float d = r[c] - m; v += d * d; }
    v *= (1.f / CC);
    float rs = rsqrtf(v + 1e-5f);
    if (tid < GRP) {
      size_t g = (size_t)(b * NN + sel[tid]);
      for (int c = 0; c < CC; ++c) {
        float y = (r[c] - m) * rs * fnw[c] + fnb[c];
        atomicAdd(&out[g * 512 + c], y);
      }
    } else {
      float* dst = cfx + (size_t)(b * SGP + p) * CC;
      for (int c = 0; c < CC; ++c) dst[c] = (r[c] - m) * rs * fnw[c] + fnb[c];
    }
  }
}

// ---------------- global stack: LayerNorm ----------------
__global__ void gk_ln(const float* __restrict__ x, const float* __restrict__ w,
                      const float* __restrict__ bia, _Float16* __restrict__ oh,
                      float* __restrict__ of) {
  __shared__ float red[256];
  int row = blockIdx.x, t = threadIdx.x;
  float v = x[(size_t)row * CC + t];
  red[t] = v; __syncthreads();
  for (int s = 128; s > 0; s >>= 1) { if (t < s) red[t] += red[t + s]; __syncthreads(); }
  float mean = red[0] * (1.f / CC);
  __syncthreads();
  float d = v - mean;
  red[t] = d * d; __syncthreads();
  for (int s = 128; s > 0; s >>= 1) { if (t < s) red[t] += red[t + s]; __syncthreads(); }
  float var = red[0] * (1.f / CC);
  float y = d * rsqrtf(var + 1e-5f) * w[t] + bia[t];
  if (oh) oh[(size_t)row * CC + t] = (_Float16)y;
  if (of) of[(size_t)row * CC + t] = y;
}

// ---------------- global stack: GEMM  C = A W^T (+bias); A tile staged in LDS ----------------
__global__ void gk_gemm(const _Float16* __restrict__ A, const _Float16* __restrict__ W,
                        const float* __restrict__ bias, _Float16* __restrict__ outh,
                        float* __restrict__ outf, int O, int mode) {
  __shared__ _Float16 as[16 * CC];   // 8KB A row-tile shared by all 8 waves
  int lane = threadIdx.x & 31, wv = threadIdx.x >> 5;
  int rt = blockIdx.x;
  int ct = blockIdx.y * 8 + wv;
  // stage A tile (coalesced b128)
  {
    const float4* src = (const float4*)(A + (size_t)rt * 16 * CC);
    float4* dst = (float4*)as;
    for (int i = threadIdx.x; i < 16 * CC * 2 / 16; i += 256) dst[i] = src[i];
  }
  __syncthreads();
  v8f acc = {};
#pragma unroll
  for (int kb = 0; kb < 8; ++kb) {
    v16h a  = frag_row16(as, CC, 0, kb * 32, lane);
    v16h bb = frag_row16(W, CC, ct * 16, kb * 32, lane);
    acc = wmma32(a, bb, acc);
  }
  int n = ct * 16 + (lane & 15), mb = rt * 16 + ((lane & 16) ? 8 : 0);
  float bv = bias[n];
  if (mode == 0) {
#pragma unroll
    for (int v = 0; v < 8; ++v) outh[(size_t)(mb + v) * O + n] = (_Float16)(acc[v] + bv);
  } else {
#pragma unroll
    for (int v = 0; v < 8; ++v) outf[(size_t)(mb + v) * O + n] += acc[v] + bv;
  }
}

// ---------------- global stack: V transpose  vtg[s][h][d][token] ----------------
__global__ void k_transv(const _Float16* __restrict__ qkvg, _Float16* __restrict__ vtg) {
  int e = blockIdx.x * 256 + threadIdx.x;
  if (e >= BB * HEADS * 64 * SGP) return;
  int t = e % SGP;
  int rest = e / SGP;
  int d = rest & 63, h = (rest >> 6) & 3, s = rest >> 8;
  vtg[e] = qkvg[(size_t)(s * SGP + t) * 768 + 512 + h * 64 + d];
}

// ---------------- global stack: attention (one 16-row query tile per block) ----------------
__global__ void gk_attn(const _Float16* __restrict__ qkv, const _Float16* __restrict__ vtg,
                        _Float16* __restrict__ obuf) {
  __shared__ float    sr[16 * SGP];
  __shared__ _Float16 ar[16 * SGP];
  __shared__ _Float16 qs[16 * 64];    // staged Q tile
  int blk = blockIdx.x;               // s*136 + h*34 + rt
  int s = blk / 136, r = blk % 136, h = r / 34, rt = r % 34;
  const _Float16* base = qkv + (size_t)s * SGP * 768;
  int lane = threadIdx.x & 31, wv = threadIdx.x >> 5;

  // stage Q tile rows rt*16..+15, cols h*64..+63
  for (int i = threadIdx.x; i < 128; i += 256) {
    int rr = i >> 3, j = i & 7;
    ((float4*)(qs + rr * 64))[j] =
        ((const float4*)(base + (size_t)(rt * 16 + rr) * 768 + h * 64))[j];
  }
  __syncthreads();

  // scores (16 x 544)
  for (int nt = wv; nt < 34; nt += 8) {
    v8f acc = {};
#pragma unroll
    for (int kb = 0; kb < 2; ++kb) {
      v16h a  = frag_row16(qs, 64, 0, kb * 32, lane);
      v16h bb = frag_row16(base + 256 + h * 64, 768, nt * 16, kb * 32, lane);
      acc = wmma32(a, bb, acc);
    }
    int n = nt * 16 + (lane & 15), mb = ((lane & 16) ? 8 : 0);
#pragma unroll
    for (int v = 0; v < 8; ++v) sr[(mb + v) * SGP + n] = acc[v] * 0.125f;
  }
  __syncthreads();

  // masked softmax over 513 valid keys
  if (threadIdx.x < 16) {
    float* srow = sr + threadIdx.x * SGP;
    float m = -1e30f;
    for (int j = 0; j < 513; ++j) m = fmaxf(m, srow[j]);
    float ss = 0.f;
    for (int j = 0; j < 513; ++j) ss += __expf(srow[j] - m);
    float inv = 1.f / ss;
    _Float16* arow = ar + threadIdx.x * SGP;
    for (int j = 0; j < 513; ++j) arow[j] = (_Float16)(__expf(srow[j] - m) * inv);
    for (int j = 513; j < SGP; ++j) arow[j] = (_Float16)0.f;
  }
  __syncthreads();

  // O = A V : 4 col tiles, K = 544 = 17*32 ; B from vtg (K-contiguous b128 loads)
  const _Float16* vb = vtg + (size_t)(s * HEADS + h) * 64 * SGP;
  for (int ct = wv; ct < 4; ct += 8) {
    v8f acc = {};
    for (int kb = 0; kb < 17; ++kb) {
      v16h a  = frag_row16(ar, SGP, 0, kb * 32, lane);
      v16h bb = frag_row16(vb, SGP, ct * 16, kb * 32, lane);
      acc = wmma32(a, bb, acc);
    }
    _Float16* od = obuf + (size_t)(s * SGP + rt * 16) * CC;
    tileStoreF16(od, CC, 0, h * 64 + ct * 16, acc, lane);
  }
}

// ---------------- finalize: rec mean + residual, closest-cluster gather ----------------
__global__ void k_finalize(float* __restrict__ out, const float* __restrict__ pdata,
                           const float* __restrict__ cnt, const int* __restrict__ closest,
                           const float* __restrict__ cfn) {
  int row = blockIdx.x;               // b*N + n
  int c = threadIdx.x;
  size_t o = (size_t)row * 512;
  float acc = out[o + c];
  float cv = fmaxf(cnt[row], 1.0f);
  out[o + c] = pdata[(size_t)row * CC + c] + acc / cv;
  int b = row >> 14;
  int cl = closest[row];
  out[o + 256 + c] = cfn[((size_t)b * SGP + cl) * CC + c];
}

__global__ void k_gtoken(float* __restrict__ out, const float* __restrict__ cfn) {
  int b = blockIdx.x, c = threadIdx.x;
  out[(size_t)BB * NN * 512 + b * CC + c] = cfn[((size_t)b * SGP + NP) * CC + c];
}

// ---------------- host launcher ----------------
extern "C" void kernel_launch(void* const* d_in, const int* in_sizes, int n_in,
                              void* d_out, int out_size, void* d_ws, size_t ws_size,
                              hipStream_t stream) {
  const float* xyz   = (const float*)d_in[0];
  const float* pdata = (const float*)d_in[1];
  const int*   fps   = (const int*)d_in[2];
  const float* l_ln1w = (const float*)d_in[3];
  const float* l_ln1b = (const float*)d_in[4];
  const float* l_qkvw = (const float*)d_in[5];
  const float* l_qkvb = (const float*)d_in[6];
  const float* l_outw = (const float*)d_in[7];
  const float* l_outb = (const float*)d_in[8];
  const float* l_ln2w = (const float*)d_in[9];
  const float* l_ln2b = (const float*)d_in[10];
  const float* l_ffnw = (const float*)d_in[11];
  const float* l_ffnb = (const float*)d_in[12];
  const float* l_fnw  = (const float*)d_in[13];
  const float* l_fnb  = (const float*)d_in[14];
  const float* g_ln1w = (const float*)d_in[15];
  const float* g_ln1b = (const float*)d_in[16];
  const float* g_qkvw = (const float*)d_in[17];
  const float* g_qkvb = (const float*)d_in[18];
  const float* g_outw = (const float*)d_in[19];
  const float* g_outb = (const float*)d_in[20];
  const float* g_ln2w = (const float*)d_in[21];
  const float* g_ln2b = (const float*)d_in[22];
  const float* g_ffnw = (const float*)d_in[23];
  const float* g_ffnb = (const float*)d_in[24];
  const float* g_fnw  = (const float*)d_in[25];
  const float* g_fnb  = (const float*)d_in[26];
  float* outp = (float*)d_out;

  // workspace carve
  char* wsb = (char*)d_ws;
  size_t off = 0;
  auto alloc = [&](size_t bytes) -> char* {
    char* p = wsb + off;
    off += (bytes + 255) & ~(size_t)255;
    return p;
  };
  _Float16* qkvw_l = (_Float16*)alloc((size_t)LAYERS * 768 * CC * 2);
  _Float16* outw_l = (_Float16*)alloc((size_t)LAYERS * CC * CC * 2);
  _Float16* ffnw_l = (_Float16*)alloc((size_t)LAYERS * CC * CC * 2);
  _Float16* qkvw_g = (_Float16*)alloc((size_t)LAYERS * 768 * CC * 2);
  _Float16* outw_g = (_Float16*)alloc((size_t)LAYERS * CC * CC * 2);
  _Float16* ffnw_g = (_Float16*)alloc((size_t)LAYERS * CC * CC * 2);
  int*      idx    = (int*)alloc((size_t)BB * NP * GRP * 4);
  float*    cent   = (float*)alloc((size_t)BB * NP * 3 * 4);
  int*      clst   = (int*)alloc((size_t)BB * NN * 4);
  float*    cnt    = (float*)alloc((size_t)BB * NN * 4);
  float*    cfx    = (float*)alloc((size_t)RG * CC * 4);
  _Float16* nbg    = (_Float16*)alloc((size_t)RG * CC * 2);
  _Float16* qkvg   = (_Float16*)alloc((size_t)RG * 768 * 2);
  _Float16* obg    = (_Float16*)alloc((size_t)RG * CC * 2);
  _Float16* vtg    = (_Float16*)alloc((size_t)BB * HEADS * 64 * SGP * 2);
  float*    cfn    = (float*)alloc((size_t)RG * CC * 4);

  // deterministic zero-init (ws/d_out are NOT re-poisoned between replays)
  hipMemsetAsync(d_out, 0, (size_t)out_size * sizeof(float), stream);
  hipMemsetAsync(cnt, 0, (size_t)BB * NN * 4, stream);
  hipMemsetAsync(cfx, 0, (size_t)RG * CC * 4, stream);

  // weight f32 -> f16
  k_cvt<<<1024, 256, 0, stream>>>(l_qkvw, qkvw_l, LAYERS * 768 * CC);
  k_cvt<<<1024, 256, 0, stream>>>(l_outw, outw_l, LAYERS * CC * CC);
  k_cvt<<<1024, 256, 0, stream>>>(l_ffnw, ffnw_l, LAYERS * CC * CC);
  k_cvt<<<1024, 256, 0, stream>>>(g_qkvw, qkvw_g, LAYERS * 768 * CC);
  k_cvt<<<1024, 256, 0, stream>>>(g_outw, outw_g, LAYERS * CC * CC);
  k_cvt<<<1024, 256, 0, stream>>>(g_ffnw, ffnw_g, LAYERS * CC * CC);

  // grouping
  k_topk<<<BB * NP, 256, 68352, stream>>>(xyz, fps, pdata, idx, cent);
  k_closest<<<dim3(NN / 256, BB), 256, 0, stream>>>(xyz, cent, clst);
  k_cnt<<<BB * NP * GRP / 256, 256, 0, stream>>>(idx, cnt);

  // local transformer (scatters pts into d_out, cluster tokens into cfx)
  k_local<<<BB * NP, 256, LOC_SMEM, stream>>>(
      pdata, idx, qkvw_l, outw_l, ffnw_l,
      l_ln1w, l_ln1b, l_qkvb, l_outb, l_ln2w, l_ln2b, l_ffnb, l_fnw, l_fnb,
      outp, cfx);

  // global transformer
  for (int l = 0; l < LAYERS; ++l) {
    gk_ln<<<RG, 256, 0, stream>>>(cfx, g_ln1w + l * CC, g_ln1b + l * CC, nbg, nullptr);
    gk_gemm<<<dim3(RG / 16, 6), 256, 0, stream>>>(nbg, qkvw_g + (size_t)l * 768 * CC,
                                                  g_qkvb + l * 768, qkvg, nullptr, 768, 0);
    k_transv<<<(BB * HEADS * 64 * SGP + 255) / 256, 256, 0, stream>>>(qkvg, vtg);
    gk_attn<<<BB * HEADS * (SGP / 16), 256, 0, stream>>>(qkvg, vtg, obg);
    gk_gemm<<<dim3(RG / 16, 2), 256, 0, stream>>>(obg, outw_g + (size_t)l * CC * CC,
                                                  g_outb + l * CC, nullptr, cfx, CC, 1);
    gk_ln<<<RG, 256, 0, stream>>>(cfx, g_ln2w + l * CC, g_ln2b + l * CC, nbg, nullptr);
    gk_gemm<<<dim3(RG / 16, 2), 256, 0, stream>>>(nbg, ffnw_g + (size_t)l * CC * CC,
                                                  g_ffnb + l * CC, nullptr, cfx, CC, 1);
  }
  gk_ln<<<RG, 256, 0, stream>>>(cfx, g_fnw, g_fnb, nbg, cfn);

  // assembly
  k_finalize<<<BB * NN, 256, 0, stream>>>(outp, pdata, cnt, clst, cfn);
  k_gtoken<<<BB, 256, 0, stream>>>(outp, cfn);
  (void)in_sizes; (void)n_in; (void)ws_size;
}
